// GreedyMerging_87428354277730
// MI455X (gfx1250) — matrix operand order repaired and checked by reading, not compiled, not run on hardware
//
#include <hip/hip_runtime.h>
#include <hip/hip_bf16.h>
#include <math.h>

typedef __attribute__((ext_vector_type(16))) _Float16 v16h;
typedef __attribute__((ext_vector_type(8)))  _Float16 v8h;
typedef __attribute__((ext_vector_type(8)))  float    v8f;

#define NTOK 256
#define NOUT 128
#define DDIM 64
#define KN_STRIDE 72                     // halves per row (144B -> conflict-free)
#define KN_BYTES (NTOK * KN_STRIDE * 2)  // 36864 bytes per array
#define NTILES 136                       // 16*17/2 upper-triangular 16x16 tiles
#define NTHREADS 256

__device__ __forceinline__ int tile_id(int ta, int tb) {
    // enumeration: ta rows ascending, tb = ta..15 ascending
    return tb + ((ta * (31 - ta)) >> 1);
}

__global__ __launch_bounds__(NTHREADS)
void greedy_merge_kernel(const float* __restrict__ Kin,
                         const float* __restrict__ Vin,
                         float* __restrict__ Out)
{
    extern __shared__ char dynsmem[];
    _Float16* khi = (_Float16*)dynsmem;                  // 256 x 72 halves (hi)
    _Float16* klo = (_Float16*)(dynsmem + KN_BYTES);     // 256 x 72 halves (lo)
    float*    Spk = (float*)(dynsmem + 2 * KN_BYTES);    // 136 tiles x 256 f32

    __shared__ int   gid[NTOK];
    __shared__ int   cnt[NTOK];
    __shared__ int   slotmap[NTOK];
    __shared__ int   gstart[NOUT];
    __shared__ int   gcnt[NOUT];
    __shared__ int   members[NTOK];
    __shared__ float redS[NTHREADS];
    __shared__ int   redC[NTHREADS];
    __shared__ int   sel_i, sel_j;

    const int t    = threadIdx.x;
    const int bb   = blockIdx.x;
    const int lane = t & 31;
    const int wave = t >> 5;

    // ---- Phase A: normalize k row t (f32), store hi/lo f16 split in LDS ----
    {
        const float* krow = Kin + ((size_t)bb * NTOK + t) * DDIM;
        float ss = 0.f;
        #pragma unroll 8
        for (int d = 0; d < DDIM; ++d) { float x = krow[d]; ss += x * x; }
        float rn = 1.0f / sqrtf(ss);
        _Float16* dh = khi + t * KN_STRIDE;
        _Float16* dl = klo + t * KN_STRIDE;
        #pragma unroll 8
        for (int d = 0; d < DDIM; ++d) {
            float x = krow[d] * rn;
            _Float16 h = (_Float16)x;
            dh[d] = h;
            dl[d] = (_Float16)(x - (float)h);
        }
        gid[t] = t;
    }
    __syncthreads();

    // ---- Phase B1: Gram matrix via split-f16 WMMA (f32-class accuracy) ----
    // S ~= hi*hi^T + hi*lo^T + lo*hi^T ; 136 tiles round-robin over 8 waves.
    {
        int tileIdx = wave;
        while (tileIdx < NTILES) {
            int ti = 0, rem = tileIdx;
            while (rem >= 16 - ti) { rem -= 16 - ti; ++ti; }
            const int I = ti * 16;
            const int J = (ti + rem) * 16;

            const int m    = lane & 15;
            const int koff = (lane >> 4) * 8;    // A: 8-half chunk base
            const int kb   = (lane >> 4) * 8;    // B: 8-half chunk base
            const int ra   = (I + m) * KN_STRIDE;
            const int rb   = (J + m) * KN_STRIDE;

            v8f c = {};
            #pragma unroll
            for (int kk = 0; kk < DDIM; kk += 32) {
                v8h ahl = *(const v8h*)(khi + ra + kk + koff);
                v8h ahh = *(const v8h*)(khi + ra + kk + koff + 16);
                v8h all = *(const v8h*)(klo + ra + kk + koff);
                v8h alh = *(const v8h*)(klo + ra + kk + koff + 16);
                v8h bhl = *(const v8h*)(khi + rb + kk + kb);
                v8h bhh = *(const v8h*)(khi + rb + kk + kb + 16);
                v8h bll = *(const v8h*)(klo + rb + kk + kb);
                v8h blh = *(const v8h*)(klo + rb + kk + kb + 16);
                v16h Ah, Al, Bh, Bl;
                #pragma unroll
                for (int q = 0; q < 8; ++q) {
                    Ah[q] = ahl[q]; Ah[q + 8] = ahh[q];
                    Al[q] = all[q]; Al[q + 8] = alh[q];
                    Bh[q] = bhl[q]; Bh[q + 8] = bhh[q];
                    Bl[q] = bll[q]; Bl[q + 8] = blh[q];
                }
                c = __builtin_amdgcn_wmma_f32_16x16x32_f16(
                        false, Ah, false, Bh, (short)0, c, false, false);
                c = __builtin_amdgcn_wmma_f32_16x16x32_f16(
                        false, Ah, false, Bl, (short)0, c, false, false);
                c = __builtin_amdgcn_wmma_f32_16x16x32_f16(
                        false, Al, false, Bh, (short)0, c, false, false);
            }
            const int mrow = (lane >> 4) * 8;
            float* tdst = Spk + tileIdx * 256;
            #pragma unroll
            for (int r = 0; r < 8; ++r)
                tdst[(r + mrow) * 16 + m] = c[r];
            tileIdx += 8;
        }
    }
    __syncthreads();

    // ---- Phase B2: greedy merge, 128 masked-argmax steps over upper tri ----
    for (int it = 0; it < NTOK - NOUT; ++it) {
        float bestS = -INFINITY;
        int   bestC = 0x7fffffff;            // tie-break key: a*256+b (edge order)
        const int gt = gid[t];
        for (int k = 0; k < NTOK; ++k) {
            if (k == t) continue;
            int gk = gid[k];
            if (gk == gt) continue;
            int a = (k < t) ? k : t;
            int b = (k < t) ? t : k;
            int addr = (tile_id(a >> 4, b >> 4) << 8) + ((a & 15) << 4) + (b & 15);
            float s = Spk[addr];
            int key = (a << 8) + b;
            if (s > bestS || (s == bestS && key < bestC)) { bestS = s; bestC = key; }
        }
        redS[t] = bestS; redC[t] = bestC;
        __syncthreads();
        for (int off = NTHREADS >> 1; off > 0; off >>= 1) {
            if (t < off) {
                float s2 = redS[t + off]; int c2 = redC[t + off];
                if (s2 > redS[t] || (s2 == redS[t] && c2 < redC[t])) {
                    redS[t] = s2; redC[t] = c2;
                }
            }
            __syncthreads();
        }
        if (t == 0) {
            int c = redC[0];
            int a = c >> 8, b = c & 255;
            sel_i = gid[a];          // group of smaller token index dies
            sel_j = gid[b];          // group of larger token index survives
        }
        __syncthreads();
        if (gid[t] == sel_i) gid[t] = sel_j;
        __syncthreads();
    }

    // ---- Phase C: deterministic group gather + output ----
    cnt[t] = 0;
    __syncthreads();
    atomicAdd(&cnt[gid[t]], 1);
    __syncthreads();
    if (t == 0) {
        int s = 0, pos = 0;
        for (int p = 0; p < NTOK; ++p) {
            if (cnt[p] > 0) {
                slotmap[p] = s;
                gstart[s]  = pos;
                gcnt[s]    = cnt[p];
                pos += cnt[p];
                ++s;
            }
        }
    }
    __syncthreads();
    {   // member lists in deterministic (token-index) order
        int g = gid[t], r = 0;
        for (int y = 0; y < t; ++y) if (gid[y] == g) ++r;
        members[gstart[slotmap[g]] + r] = t;
    }
    __syncthreads();
    {   // 2 threads per output row: each accumulates a 32-wide half-row
        const int o     = t >> 1;
        const int dbase = (t & 1) * 32;
        const int st    = gstart[o];
        const int n     = gcnt[o];
        const float inv = 1.0f / (float)n;
        float acc[32];
        #pragma unroll
        for (int d = 0; d < 32; ++d) acc[d] = 0.f;
        for (int q = 0; q < n; ++q) {
            const float* vrow =
                Vin + ((size_t)bb * NTOK + members[st + q]) * DDIM + dbase;
            #pragma unroll 8
            for (int d = 0; d < 32; ++d) acc[d] += vrow[d];
        }
        float* orow = Out + ((size_t)bb * NOUT + o) * DDIM + dbase;
        #pragma unroll 8
        for (int d = 0; d < 32; ++d) orow[d] = acc[d] * inv;
        if ((t & 1) == 0) {
            float* sizes = Out + (size_t)gridDim.x * NOUT * DDIM;
            sizes[(size_t)bb * NOUT + o] = (float)n;
        }
    }
}

extern "C" void kernel_launch(void* const* d_in, const int* in_sizes, int n_in,
                              void* d_out, int out_size, void* d_ws, size_t ws_size,
                              hipStream_t stream) {
    (void)n_in; (void)out_size; (void)d_ws; (void)ws_size;
    const float* Kin = (const float*)d_in[0];
    const float* Vin = (const float*)d_in[1];
    float* Out = (float*)d_out;
    const int B = in_sizes[0] / (NTOK * DDIM);   // 128
    const size_t shbytes = 2 * (size_t)KN_BYTES
                         + (size_t)NTILES * 256 * sizeof(float);  // ~208 KB
    greedy_merge_kernel<<<B, NTHREADS, shbytes, stream>>>(Kin, Vin, Out);
}
